// TransformerPlanner_82231443849776
// MI455X (gfx1250) — matrix operand order repaired and checked by reading, not compiled, not run on hardware
//
#include <hip/hip_runtime.h>
#include <hip/hip_bf16.h>

// ---------------------------------------------------------------------------
// TransformerPlanner on gfx1250: per-batch micro-transformer.
// All GEMMs use v_wmma_f32_16x16x32_f16 (f16 inputs, f32 accumulate).
// Prep kernels hoist batch-invariant work (query LN + q-projection) and
// convert weights to f16 once per launch; weights stay hot in the 192MB L2.
// All WMMA D-writebacks are branch-free (padded LDS rows, padded qln).
// ---------------------------------------------------------------------------

typedef __attribute__((ext_vector_type(16))) _Float16 v16h;
typedef __attribute__((ext_vector_type(8)))  _Float16 v8h;
typedef __attribute__((ext_vector_type(8)))  float    v8f;

#define SHUF16(lo, hi) __builtin_shufflevector((lo), (hi), 0,1,2,3,4,5,6,7,8,9,10,11,12,13,14,15)

__device__ __forceinline__ float wave_sum(float v) {
#pragma unroll
  for (int m = 16; m >= 1; m >>= 1) v += __shfl_xor(v, m, 32);
  return v;
}
__device__ __forceinline__ float wave_max(float v) {
#pragma unroll
  for (int m = 16; m >= 1; m >>= 1) v = fmaxf(v, __shfl_xor(v, m, 32));
  return v;
}

// A-operand loader: 16x32 f16 tile, contiguous layout.
// lane group g = lane>>4 ; row M = (lane&15)+tile_row_base.
// halves 0..7  = K = koff + g*16 + 0..7
// halves 8..15 = K = koff + g*16 + 8..15
__device__ __forceinline__ v16h load_a_tile(const _Float16* rowPtr, int koff, int g) {
  v8h lo = *(const v8h*)(rowPtr + koff + g * 16);
  v8h hi = *(const v8h*)(rowPtr + koff + g * 16 + 8);
  return SHUF16(lo, hi);
}

// ---------------------------------------------------------------------------
// Prep 1: fp32 -> f16 weight conversion (grid-stride).
// wk (in_proj rows 128..255), wv (rows 256..383), out_proj, fc1, fc2.
// ---------------------------------------------------------------------------
__global__ __launch_bounds__(256) void planner_prep_weights(
    const float* __restrict__ in_proj_w, const float* __restrict__ out_proj_w,
    const float* __restrict__ fc1_w, const float* __restrict__ fc2_w,
    _Float16* __restrict__ wk_h, _Float16* __restrict__ wv_h,
    _Float16* __restrict__ wo_h, _Float16* __restrict__ fc1_h,
    _Float16* __restrict__ fc2_h) {
  const int total = 16384 * 3 + 32768 * 2; // 114688
  for (int i = blockIdx.x * blockDim.x + threadIdx.x; i < total;
       i += gridDim.x * blockDim.x) {
    if (i < 16384) {
      wk_h[i] = (_Float16)in_proj_w[16384 + i];
    } else if (i < 32768) {
      int j = i - 16384;
      wv_h[j] = (_Float16)in_proj_w[32768 + j];
    } else if (i < 49152) {
      int j = i - 32768;
      wo_h[j] = (_Float16)out_proj_w[j];
    } else if (i < 81920) {
      int j = i - 49152;
      fc1_h[j] = (_Float16)fc1_w[j];
    } else {
      int j = i - 81920;
      fc2_h[j] = (_Float16)fc2_w[j];
    }
  }
}

// ---------------------------------------------------------------------------
// Prep 2: batch-invariant queries: qln = LN(query_emb), q = qln @ Wq^T + bq.
// qln is padded to 16 rows (rows 3..15 zero) so the main kernel's residual
// add is branch-free. One block of 128 threads.
// ---------------------------------------------------------------------------
__global__ __launch_bounds__(128) void planner_prep_q(
    const float* __restrict__ query_emb, const float* __restrict__ qg,
    const float* __restrict__ qb, const float* __restrict__ wq,
    const float* __restrict__ bq, float* __restrict__ q_f32,
    float* __restrict__ qln_pad) {
  __shared__ float red[4];
  __shared__ float qs[3][128];
  const int t = threadIdx.x, lane = t & 31, w32 = t >> 5;
  // zero padded rows 3..15 of qln
  for (int i = t + 384; i < 16 * 128; i += 128) qln_pad[i] = 0.f;
  for (int w = 0; w < 3; ++w) {
    float v = query_emb[w * 128 + t];
    float s = wave_sum(v);
    if (lane == 0) red[w32] = s;
    __syncthreads();
    float m = (red[0] + red[1] + red[2] + red[3]) * (1.f / 128.f);
    __syncthreads();
    float dd = v - m;
    float s2 = wave_sum(dd * dd);
    if (lane == 0) red[w32] = s2;
    __syncthreads();
    float var = (red[0] + red[1] + red[2] + red[3]) * (1.f / 128.f);
    __syncthreads();
    float xn = dd * rsqrtf(var + 1e-5f) * qg[t] + qb[t];
    qs[w][t] = xn;
    qln_pad[w * 128 + t] = xn;
    __syncthreads();
  }
  for (int idx = t; idx < 384; idx += 128) {
    int w = idx >> 7, n = idx & 127;
    float o = bq[n];
#pragma unroll 8
    for (int d = 0; d < 128; ++d) o += qs[w][d] * wq[n * 128 + d];
    q_f32[idx] = o;
  }
}

// ---------------------------------------------------------------------------
// Main kernel: one batch per 256-thread block (8 wave32s).
// ---------------------------------------------------------------------------
struct PParams {
  const float* track_left;
  const float* track_right;
  const float* coord_w;
  const float* coord_b;
  const float* pos_emb;
  const float* side_emb;
  const float* tln_g;
  const float* tln_b;
  const float* in_proj_b;
  const float* out_proj_b;
  const float* fc1_b;
  const float* fc2_b;
  const float* paln_g;
  const float* paln_b;
  const float* pfln_g;
  const float* pfln_b;
  const float* head_w;
  const float* head_b;
  const _Float16* wk_h;
  const _Float16* wv_h;
  const _Float16* wo_h;
  const _Float16* fc1_h;
  const _Float16* fc2_h;
  const float* q_f32;
  const float* qln_pad;
  float* out;
};

__global__ __launch_bounds__(256) void planner_main(PParams P) {
  __shared__ __align__(16) _Float16 tok_h[32][136]; // tokens f16, padded
  // fbufA: kbuf (32x132, rows 20..31 garbage) then y32 (rows 0..15)
  // fbufB: vbuf (32x132)                      then x32 (rows 0..15)
  __shared__ __align__(16) float fbufA[32][132];
  __shared__ __align__(16) float fbufB[32][132];
  __shared__ float attnS[8][3][20];
  __shared__ __align__(16) _Float16 ctx_h[16][136];
  __shared__ __align__(16) _Float16 xf16[16][136];
  __shared__ __align__(16) _Float16 hf16[16][264];

  const int tid = threadIdx.x;
  const int wave = tid >> 5, lane = tid & 31;
  const int g = lane >> 4, col = lane & 15;
  const int b = blockIdx.x;

  // ---- zero the f16 staging buffers (padding rows/cols must be 0) ----
  for (int i = tid; i < 32 * 136; i += 256) (&tok_h[0][0])[i] = (_Float16)0.f;
  for (int i = tid; i < 16 * 136; i += 256) {
    (&ctx_h[0][0])[i] = (_Float16)0.f;
    (&xf16[0][0])[i] = (_Float16)0.f;
  }
  __syncthreads();

  // ---- Step 1: tokens = LN(coord_feat + pos + side), one wave per row ----
  for (int r = wave; r < 20; r += 8) {
    const int t = r % 10;
    const int side = r / 10;
    const float* trk = side ? P.track_right : P.track_left;
    const float cx = trk[((size_t)b * 10 + t) * 2 + 0];
    const float cy = trk[((size_t)b * 10 + t) * 2 + 1];
    float vals[4];
    float s1 = 0.f, s2 = 0.f;
    const int d0 = lane * 4;
#pragma unroll
    for (int j = 0; j < 4; ++j) {
      int d = d0 + j;
      float vv = cx * P.coord_w[2 * d] + cy * P.coord_w[2 * d + 1] +
                 P.coord_b[d] + P.pos_emb[t * 128 + d] +
                 P.side_emb[side * 128 + d];
      vals[j] = vv;
      s1 += vv;
      s2 += vv * vv;
    }
    s1 = wave_sum(s1);
    s2 = wave_sum(s2);
    float m = s1 * (1.f / 128.f);
    float var = s2 * (1.f / 128.f) - m * m;
    float rs = rsqrtf(var + 1e-5f);
#pragma unroll
    for (int j = 0; j < 4; ++j) {
      int d = d0 + j;
      float xn = (vals[j] - m) * rs * P.tln_g[d] + P.tln_b[d];
      tok_h[r][d] = (_Float16)xn;
    }
  }
  __syncthreads();

  // ---- Step 2: K and V projections (20x128 @ 128x128), WMMA ----
  {
    const int nt = wave;          // each wave owns one 16-col tile
    const int n = nt * 16 + col;  // output column
#pragma unroll
    for (int p = 0; p < 2; ++p) {
      const _Float16* Wh = p ? P.wv_h : P.wk_h;
      const float bias = P.in_proj_b[128 + p * 128 + n];
      v8f acc0 = {};
      v8f acc1 = {};
#pragma unroll
      for (int ks = 0; ks < 4; ++ks) {
        const int koff = ks * 32;
        v16h bm = *(const v16h*)(Wh + n * 128 + koff + g * 16);
        v16h a0 = load_a_tile(&tok_h[col][0], koff, g);
        v16h a1 = load_a_tile(&tok_h[16 + col][0], koff, g);
        acc0 = __builtin_amdgcn_wmma_f32_16x16x32_f16(false, a0, false, bm,
                                                      (short)0, acc0, false, false);
        acc1 = __builtin_amdgcn_wmma_f32_16x16x32_f16(false, a1, false, bm,
                                                      (short)0, acc1, false, false);
      }
      float(*obuf)[132] = p ? fbufB : fbufA;
#pragma unroll
      for (int j = 0; j < 8; ++j) { // branch-free: rows 20..31 are dead pads
        obuf[g * 8 + j][n] = acc0[j] + bias;
        obuf[16 + g * 8 + j][n] = acc1[j] + bias;
      }
    }
  }
  __syncthreads();

  // ---- Step 3: scores + softmax; one wave per head, lane = key index ----
  {
    const int h = wave;
    const int kk = lane;
#pragma unroll
    for (int w = 0; w < 3; ++w) {
      float s;
      if (kk < 20) {
        s = 0.f;
#pragma unroll
        for (int j = 0; j < 16; ++j)
          s += P.q_f32[w * 128 + h * 16 + j] * fbufA[kk][h * 16 + j];
        s *= 0.25f; // 1/sqrt(16)
      } else {
        s = -__builtin_inff();
      }
      float mx = wave_max(s);
      float e = (kk < 20) ? __expf(s - mx) : 0.f;
      float sum = wave_sum(e);
      if (kk < 20) attnS[h][w][kk] = e / sum;
    }
  }
  __syncthreads();

  // ---- Step 4: ctx = attn @ v  (3x128), store f16 padded to 16 rows ----
  for (int idx = tid; idx < 384; idx += 256) {
    int w = idx >> 7, d = idx & 127, h = d >> 4;
    float c = 0.f;
#pragma unroll
    for (int kk = 0; kk < 20; ++kk) c += attnS[h][w][kk] * fbufB[kk][d];
    ctx_h[w][d] = (_Float16)c;
  }
  __syncthreads();

  // ---- Step 5: attn_out = ctx @ Wo^T + bo; x = qln + attn_out ----
  // fbufB (vbuf) is dead now; reuse rows 0..15 as x32.
  {
    const int n = wave * 16 + col;
    v8f acc = {};
#pragma unroll
    for (int ks = 0; ks < 4; ++ks) {
      const int koff = ks * 32;
      v16h bm = *(const v16h*)(P.wo_h + n * 128 + koff + g * 16);
      v16h a = load_a_tile(&ctx_h[col][0], koff, g);
      acc = __builtin_amdgcn_wmma_f32_16x16x32_f16(false, a, false, bm, (short)0,
                                                   acc, false, false);
    }
    const float bo = P.out_proj_b[n];
#pragma unroll
    for (int j = 0; j < 8; ++j) { // branch-free: qln rows 3..15 are zero
      int r = g * 8 + j;
      fbufB[r][n] = acc[j] + bo + P.qln_pad[r * 128 + n];
    }
  }
  __syncthreads();

  // ---- Step 6: x = LN(x) (post-attn); keep f32 residual + f16 copy ----
  if (wave < 3) {
    const int r = wave;
    float vals[4];
    float s1 = 0.f, s2 = 0.f;
    const int d0 = lane * 4;
#pragma unroll
    for (int j = 0; j < 4; ++j) {
      float vv = fbufB[r][d0 + j];
      vals[j] = vv;
      s1 += vv;
      s2 += vv * vv;
    }
    s1 = wave_sum(s1);
    s2 = wave_sum(s2);
    float m = s1 * (1.f / 128.f);
    float var = s2 * (1.f / 128.f) - m * m;
    float rs = rsqrtf(var + 1e-5f);
#pragma unroll
    for (int j = 0; j < 4; ++j) {
      int d = d0 + j;
      float xn = (vals[j] - m) * rs * P.paln_g[d] + P.paln_b[d];
      fbufB[r][d] = xn;
      xf16[r][d] = (_Float16)xn;
    }
  }
  __syncthreads();

  // ---- Step 7: h = relu(x @ fc1^T + b1)  (3x256) ----
  {
#pragma unroll
    for (int half = 0; half < 2; ++half) {
      const int n = (wave + half * 8) * 16 + col;
      v8f acc = {};
#pragma unroll
      for (int ks = 0; ks < 4; ++ks) {
        const int koff = ks * 32;
        v16h bm = *(const v16h*)(P.fc1_h + n * 128 + koff + g * 16);
        v16h a = load_a_tile(&xf16[col][0], koff, g);
        acc = __builtin_amdgcn_wmma_f32_16x16x32_f16(false, a, false, bm,
                                                     (short)0, acc, false, false);
      }
      const float bb = P.fc1_b[n];
#pragma unroll
      for (int j = 0; j < 8; ++j) {
        int r = g * 8 + j;
        hf16[r][n] = (_Float16)fmaxf(acc[j] + bb, 0.f); // rows>=3 dead
      }
    }
  }
  __syncthreads();

  // ---- Step 8: y = h @ fc2^T + b2 + x  (K=256 -> 8 k-steps) ----
  // fbufA (kbuf) is dead now; reuse rows 0..15 as y32.
  {
    const int n = wave * 16 + col;
    v8f acc = {};
#pragma unroll
    for (int ks = 0; ks < 8; ++ks) {
      const int koff = ks * 32;
      v16h bm = *(const v16h*)(P.fc2_h + n * 256 + koff + g * 16);
      v16h a = load_a_tile(&hf16[col][0], koff, g);
      acc = __builtin_amdgcn_wmma_f32_16x16x32_f16(false, a, false, bm, (short)0,
                                                   acc, false, false);
    }
    const float bb = P.fc2_b[n];
#pragma unroll
    for (int j = 0; j < 8; ++j) { // branch-free: rows 3..15 dead
      int r = g * 8 + j;
      fbufA[r][n] = acc[j] + bb + fbufB[r][n];
    }
  }
  __syncthreads();

  // ---- Step 9: x = LN(y) (post-ffn), final result into fbufB rows 0..2 ----
  if (wave < 3) {
    const int r = wave;
    float vals[4];
    float s1 = 0.f, s2 = 0.f;
    const int d0 = lane * 4;
#pragma unroll
    for (int j = 0; j < 4; ++j) {
      float vv = fbufA[r][d0 + j];
      vals[j] = vv;
      s1 += vv;
      s2 += vv * vv;
    }
    s1 = wave_sum(s1);
    s2 = wave_sum(s2);
    float m = s1 * (1.f / 128.f);
    float var = s2 * (1.f / 128.f) - m * m;
    float rs = rsqrtf(var + 1e-5f);
#pragma unroll
    for (int j = 0; j < 4; ++j) {
      int d = d0 + j;
      fbufB[r][d] = (vals[j] - m) * rs * P.pfln_g[d] + P.pfln_b[d];
    }
  }
  __syncthreads();

  // ---- Step 10: out = x @ head^T + head_b  (3x2) ----
  if (tid < 6) {
    const int w = tid >> 1, n = tid & 1;
    float o = P.head_b[n];
#pragma unroll 8
    for (int d = 0; d < 128; ++d) o += fbufB[w][d] * P.head_w[n * 128 + d];
    P.out[((size_t)b * 3 + w) * 2 + n] = o;
  }
}

// ---------------------------------------------------------------------------
// Host-side launcher.
// ---------------------------------------------------------------------------
extern "C" void kernel_launch(void* const* d_in, const int* in_sizes, int n_in,
                              void* d_out, int out_size, void* d_ws,
                              size_t ws_size, hipStream_t stream) {
  (void)in_sizes;
  (void)n_in;
  (void)out_size;
  (void)ws_size;

  const float* track_left = (const float*)d_in[0];
  const float* track_right = (const float*)d_in[1];
  const float* coord_w = (const float*)d_in[2];
  const float* coord_b = (const float*)d_in[3];
  const float* pos_emb = (const float*)d_in[4];
  const float* side_emb = (const float*)d_in[5];
  const float* query_emb = (const float*)d_in[6];
  const float* tokens_ln_g = (const float*)d_in[7];
  const float* tokens_ln_b = (const float*)d_in[8];
  const float* queries_ln_g = (const float*)d_in[9];
  const float* queries_ln_b = (const float*)d_in[10];
  const float* in_proj_w = (const float*)d_in[11];
  const float* in_proj_b = (const float*)d_in[12];
  const float* out_proj_w = (const float*)d_in[13];
  const float* out_proj_b = (const float*)d_in[14];
  const float* fc1_w = (const float*)d_in[15];
  const float* fc1_b = (const float*)d_in[16];
  const float* fc2_w = (const float*)d_in[17];
  const float* fc2_b = (const float*)d_in[18];
  const float* post_attn_ln_g = (const float*)d_in[19];
  const float* post_attn_ln_b = (const float*)d_in[20];
  const float* post_ffn_ln_g = (const float*)d_in[21];
  const float* post_ffn_ln_b = (const float*)d_in[22];
  const float* head_w = (const float*)d_in[23];
  const float* head_b = (const float*)d_in[24];

  // Workspace layout (needs 239104 bytes).
  char* ws = (char*)d_ws;
  _Float16* wk_h = (_Float16*)(ws + 0);
  _Float16* wv_h = (_Float16*)(ws + 32768);
  _Float16* wo_h = (_Float16*)(ws + 65536);
  _Float16* fc1_h = (_Float16*)(ws + 98304);
  _Float16* fc2_h = (_Float16*)(ws + 163840);
  float* q_f32 = (float*)(ws + 229376);
  float* qln_pad = (float*)(ws + 230912); // 16x128 f32, rows 3..15 zero

  planner_prep_weights<<<112, 256, 0, stream>>>(in_proj_w, out_proj_w, fc1_w,
                                                fc2_w, wk_h, wv_h, wo_h, fc1_h,
                                                fc2_h);
  planner_prep_q<<<1, 128, 0, stream>>>(query_emb, queries_ln_g, queries_ln_b,
                                        in_proj_w, in_proj_b, q_f32, qln_pad);

  PParams P;
  P.track_left = track_left;
  P.track_right = track_right;
  P.coord_w = coord_w;
  P.coord_b = coord_b;
  P.pos_emb = pos_emb;
  P.side_emb = side_emb;
  P.tln_g = tokens_ln_g;
  P.tln_b = tokens_ln_b;
  P.in_proj_b = in_proj_b;
  P.out_proj_b = out_proj_b;
  P.fc1_b = fc1_b;
  P.fc2_b = fc2_b;
  P.paln_g = post_attn_ln_g;
  P.paln_b = post_attn_ln_b;
  P.pfln_g = post_ffn_ln_g;
  P.pfln_b = post_ffn_ln_b;
  P.head_w = head_w;
  P.head_b = head_b;
  P.wk_h = wk_h;
  P.wv_h = wv_h;
  P.wo_h = wo_h;
  P.fc1_h = fc1_h;
  P.fc2_h = fc2_h;
  P.q_f32 = q_f32;
  P.qln_pad = qln_pad;
  P.out = (float*)d_out;

  planner_main<<<32768, 256, 0, stream>>>(P);
}
